// ComplexMeasurement_64587718198041
// MI455X (gfx1250) — compile-verified, matrix operand order checked by reading, not run on hardware
//
#include <hip/hip_runtime.h>
#include <hip/hip_bf16.h>

typedef __attribute__((ext_vector_type(16))) _Float16 v16h;
typedef __attribute__((ext_vector_type(8)))  _Float16 v8h;
typedef __attribute__((ext_vector_type(8)))  float    v8f;
typedef __attribute__((ext_vector_type(4)))  float    v4f;

#define BB 128          // batch
#define UU 200          // number of kernels
#define DD 512          // matrix dim
#define NTILE 64        // n-columns per workgroup
#define NPAD 256        // padded n (4 * NTILE)
#define IPC 8           // i-rows per K-chunk
#define NCHUNK (DD / IPC)   // 64 K-chunks
#define JPHASE 128      // j-extent staged in LDS per phase
#define LSTR 136        // padded LDS row stride (f16): 272B -> bank-conflict-free

// ---- fragment loaders -------------------------------------------------------
// A-matrix 16x32 f16 fragment (ISA 7.12.2): lane&15 = M row; lane<16 holds
// K 0..7 (v0-3) + 16..23 (v4-7); lane>=16 holds K 8..15 + 24..31.
// p is pre-offset by half*8 floats; converts f32 -> f16.
__device__ __forceinline__ v16h load_a_frag(const float* p) {
    v4f x0 = *(const v4f*)(p + 0);
    v4f x1 = *(const v4f*)(p + 4);
    v4f x2 = *(const v4f*)(p + 16);
    v4f x3 = *(const v4f*)(p + 20);
    v16h f;
#pragma unroll
    for (int t = 0; t < 4; ++t) {
        f[t]      = (_Float16)x0[t];
        f[4 + t]  = (_Float16)x1[t];
        f[8 + t]  = (_Float16)x2[t];
        f[12 + t] = (_Float16)x3[t];
    }
    return f;
}

// B-matrix 32x16 fragment from LDS row (row = one n, contiguous in k).
__device__ __forceinline__ v16h load_b_frag(const _Float16* row, int half) {
    const _Float16* p = row + half * 8;
    v8h lo = *(const v8h*)(p);
    v8h hi = *(const v8h*)(p + 16);
    v16h f;
#pragma unroll
    for (int t = 0; t < 8; ++t) { f[t] = lo[t]; f[8 + t] = hi[t]; }
    return f;
}

// ---- main GEMM: out[b,n] partial = <Ar_b, M_n> + <Ai_b, -N_n> over K-chunk --
template <int USE_WS>
__global__ __launch_bounds__(256) void cm_gemm(
    const float* __restrict__ Ar, const float* __restrict__ Ai,
    const float* __restrict__ kern, float* __restrict__ part,
    float* __restrict__ out)
{
    __shared__ _Float16 Mt[NTILE][LSTR];   // M_n tile, f16
    __shared__ _Float16 Nt[NTILE][LSTR];   // -N_n tile, f16 (pre-negated)

    const int tid  = threadIdx.x;
    const int lane = tid & 31;
    const int wave = tid >> 5;
    const int l15  = lane & 15;
    const int half = (lane >> 4) & 1;
    const int n0   = blockIdx.x * NTILE;
    const int kc   = blockIdx.y;
    const int i0   = kc * IPC;
    const int brow = wave * 16 + l15;      // this lane's A row (b)

    // B-tile builder mapping: thread -> (local n, 32-wide j slice)
    const int   cn    = tid >> 2;                  // 0..63
    const int   cj    = (tid & 3) * 32;            // j offset within phase
    const int   ng    = n0 + cn;
    const float maskf = (ng < UU) ? 1.0f : 0.0f;   // zero out padded n
    const int   ncl   = (ng < UU) ? ng : 0;        // clamped (in-bounds) n
    const float* kv   = kern + (size_t)ncl * DD * 2;   // kernel[n][d][0/1]

    v8f acc[4];
#pragma unroll
    for (int s = 0; s < 4; ++s)
#pragma unroll
        for (int r = 0; r < 8; ++r) acc[s][r] = 0.0f;

    for (int ii = 0; ii < IPC; ++ii) {
        const int i = i0 + ii;
        const float kri = maskf * kv[2 * i + 0];
        const float kii = maskf * kv[2 * i + 1];
        const float* arow = Ar + ((size_t)brow * DD + i) * DD;
        const float* airow = Ai + ((size_t)brow * DD + i) * DD;
        if (ii + 1 < IPC) {   // speculative prefetch of next i-row
            __builtin_prefetch(arow + DD, 0, 0);
            __builtin_prefetch(airow + DD, 0, 0);
        }

        for (int ph = 0; ph < DD / JPHASE; ++ph) {
            const int jbase = ph * JPHASE;
            __syncthreads();   // previous phase's LDS reads complete
            {   // build M / -N tiles for j in [jbase, jbase+128)
                const float* kp = kv + (size_t)(jbase + cj) * 2;
                _Float16 mtmp[32], ntmp[32];
#pragma unroll
                for (int t = 0; t < 32; ++t) {
                    const float krj = kp[2 * t + 0];
                    const float kij = kp[2 * t + 1];
                    mtmp[t] = (_Float16)( (kri * krj - kii * kij));
                    ntmp[t] = (_Float16)(-(kri * kij + kii * krj));
                }
#pragma unroll
                for (int c = 0; c < 4; ++c) {
                    *(v8h*)&Mt[cn][cj + c * 8] = *(const v8h*)&mtmp[c * 8];
                    *(v8h*)&Nt[cn][cj + c * 8] = *(const v8h*)&ntmp[c * 8];
                }
            }
            __syncthreads();   // tiles visible to all waves

#pragma unroll
            for (int jb = 0; jb < JPHASE / 32; ++jb) {
                const int j0 = jbase + jb * 32;
                const int jr = jb * 32;
                v16h afr = load_a_frag(arow + j0 + half * 8);
                v16h afi = load_a_frag(airow + j0 + half * 8);
#pragma unroll
                for (int s = 0; s < 4; ++s) {
                    v16h bm = load_b_frag(&Mt[s * 16 + l15][jr], half);
                    acc[s] = __builtin_amdgcn_wmma_f32_16x16x32_f16(
                        false, afr, false, bm, (short)0, acc[s], false, false);
                    v16h bn = load_b_frag(&Nt[s * 16 + l15][jr], half);
                    acc[s] = __builtin_amdgcn_wmma_f32_16x16x32_f16(
                        false, afi, false, bn, (short)0, acc[s], false, false);
                }
            }
        }
    }

    // Epilogue. C/D layout: VGPR r -> M = r + (half?8:0), N = l15.
    if (USE_WS) {
        float* pb = part + (size_t)kc * BB * NPAD;
#pragma unroll
        for (int s = 0; s < 4; ++s) {
            const int n = n0 + s * 16 + l15;
#pragma unroll
            for (int r = 0; r < 8; ++r) {
                const int b = wave * 16 + r + half * 8;
                pb[(size_t)b * NPAD + n] = acc[s][r];
            }
        }
    } else {
        #pragma unroll
        for (int s = 0; s < 4; ++s) {
            const int n = n0 + s * 16 + l15;
            if (n < UU) {
#pragma unroll
                for (int r = 0; r < 8; ++r) {
                    const int b = wave * 16 + r + half * 8;
                    atomicAdd(&out[(size_t)b * UU + n], acc[s][r]);
                }
            }
        }
    }
}

// ---- deterministic split-K reduction ---------------------------------------
__global__ void cm_reduce(const float* __restrict__ part, float* __restrict__ out) {
    const int idx = blockIdx.x * blockDim.x + threadIdx.x;
    if (idx >= BB * UU) return;
    const int b = idx / UU;
    const int n = idx % UU;
    float s = 0.0f;
#pragma unroll 8
    for (int kc = 0; kc < NCHUNK; ++kc)
        s += part[((size_t)kc * BB + b) * NPAD + n];
    out[idx] = s;
}

__global__ void cm_zero(float* __restrict__ out, int n) {
    const int idx = blockIdx.x * blockDim.x + threadIdx.x;
    if (idx < n) out[idx] = 0.0f;
}

// ---- launch -----------------------------------------------------------------
extern "C" void kernel_launch(void* const* d_in, const int* in_sizes, int n_in,
                              void* d_out, int out_size, void* d_ws, size_t ws_size,
                              hipStream_t stream) {
    (void)in_sizes; (void)n_in; (void)out_size;
    const float* Ar   = (const float*)d_in[0];
    const float* Ai   = (const float*)d_in[1];
    const float* kern = (const float*)d_in[2];
    float* out = (float*)d_out;

    const size_t need = (size_t)NCHUNK * BB * NPAD * sizeof(float);  // 8.39 MB
    dim3 grid(NPAD / NTILE, NCHUNK);   // (4 n-tiles, 64 K-chunks) = 256 WGs

    if (d_ws && ws_size >= need) {
        float* part = (float*)d_ws;
        cm_gemm<1><<<grid, 256, 0, stream>>>(Ar, Ai, kern, part, out);
        cm_reduce<<<(BB * UU + 255) / 256, 256, 0, stream>>>(part, out);
    } else {
        cm_zero<<<(BB * UU + 255) / 256, 256, 0, stream>>>(out, BB * UU);
        cm_gemm<0><<<grid, 256, 0, stream>>>(Ar, Ai, kern, nullptr, out);
    }
}
